// T3S_Network_17239998726624
// MI455X (gfx1250) — compile-verified, matrix operand order checked by reading, not run on hardware
//
#include <hip/hip_runtime.h>
#include <hip/hip_bf16.h>
#include <hip/hip_fp16.h>

// ---------------------------------------------------------------------------
// T3S triplet network for MI455X (gfx1250, wave32, WMMA).
//   B=32, T=512, E=128, NH=16, HD=8.
// Pipeline:
//   k_prep_m  : grid featurizer -> m (f16), elementwise
//   k_qkv     : m @ attn_in_w^T + b  via v_wmma_f32_16x16x32_f16 (K=128 in 4 steps)
//   k_gru     : sequential GRU, 1 WG per (seq,batch), h in LDS
//   k_attn    : flash attention per (seq,b,head,q-tile) via v_wmma_f32_16x16x4_f32
//               (QK^T: 2 chained K=4; PV: 4 chained K=4), emits row-sum partials
//   k_final   : mean-pool * out_w^T + out_b, gamma-combine -> d_out
//   k_dist    : exp(-||a-p+eps||), exp(-||a-n+eps||)
// ---------------------------------------------------------------------------

typedef __attribute__((ext_vector_type(16))) _Float16 v16h;
typedef __attribute__((ext_vector_type(8)))  float    v8f;
typedef __attribute__((ext_vector_type(2)))  float    v2f;

#define NS_ 3
#define B_  32
#define T_  512
#define E_  128
#define NH_ 16
#define HD_ 8

// ------------------------------ k_prep_m -----------------------------------
// m[s,b,t,e] = lookup(grid) + posres(grid + [sin t, cos t])    (stored f16)
__global__ void k_prep_m(const float* __restrict__ a_in,
                         const float* __restrict__ p_in,
                         const float* __restrict__ n_in,
                         const float* __restrict__ lookup_w,
                         const float* __restrict__ lookup_b,
                         const float* __restrict__ posres_w,
                         const float* __restrict__ posres_b,
                         _Float16* __restrict__ m) {
  int idx = blockIdx.x * blockDim.x + threadIdx.x;   // over NS*B*T*E
  if (idx >= NS_ * B_ * T_ * E_) return;
  int e  = idx & (E_ - 1);
  int t  = (idx >> 7) & (T_ - 1);
  int sb = idx >> 16;                 // s*B + b
  int s  = sb >> 5, b = sb & 31;
  const float* in = (s == 0) ? a_in : (s == 1) ? p_in : n_in;
  const float* g  = in + ((size_t)b * T_ + t) * 4 + 2;
  float g0 = g[0], g1 = g[1];
  float sp = sinf((float)t), cp = cosf((float)t);
  float val = lookup_w[e * 2 + 0] * g0 + lookup_w[e * 2 + 1] * g1 + lookup_b[e]
            + posres_w[e * 2 + 0] * (g0 + sp) + posres_w[e * 2 + 1] * (g1 + cp)
            + posres_b[e];
  m[idx] = (_Float16)val;
}

// ------------------------------- k_qkv -------------------------------------
// One wave per 16x16 output tile of (49152 rows x 384 cols), K=128 over 4 WMMA.
// A-frag (16-bit 16x32): lane L -> row M=L&15; halves j<8: K=32k+8*hv+j,
// j>=8: K=32k+16+8*hv+(j-8). B mirrors A with N=L&15 (B[k][n] = W[n][k]).
__global__ void __launch_bounds__(256)
k_qkv(const _Float16* __restrict__ m, const float* __restrict__ W,
      const float* __restrict__ bias,
      _Float16* __restrict__ q, _Float16* __restrict__ k,
      _Float16* __restrict__ v) {
  int wave = blockIdx.x * (blockDim.x >> 5) + (threadIdx.x >> 5);
  int lane = threadIdx.x & 31;
  int M  = lane & 15;
  int hv = lane >> 4;
  const int rowTiles = (NS_ * B_ * T_) / 16;   // 3072
  int ct = wave % 24;                          // 384/16 col tiles
  int rt = wave / 24;
  if (rt >= rowTiles) return;

  int rowA = rt * 16 + M;    // A row
  int colB = ct * 16 + M;    // B column (N)
  v8f c = {};
#pragma unroll
  for (int ks = 0; ks < 4; ++ks) {
    int kb = ks * 32 + hv * 8;
    v16h a, bf;
    const _Float16* mrow = m + (size_t)rowA * E_ + kb;
    const float*    wrow = W + (size_t)colB * E_ + kb;
#pragma unroll
    for (int j = 0; j < 8; ++j) {
      a[j]      = mrow[j];
      a[j + 8]  = mrow[16 + j];
      bf[j]     = (_Float16)wrow[j];
      bf[j + 8] = (_Float16)wrow[16 + j];
    }
    c = __builtin_amdgcn_wmma_f32_16x16x32_f16(false, a, false, bf,
                                               (short)0, c, false, false);
  }
  // Scatter C (lane: col N=L&15; VGPR j: row 8*hv+j) into q/k/v f16 buffers.
  int colAbs = ct * 16 + M;
  float bcol = bias[colAbs];
  int sect = colAbs >> 7;          // 0=q 1=k 2=v
  int hh   = (colAbs >> 3) & 15;   // head
  int dd   = colAbs & 7;           // dim in head
#pragma unroll
  for (int j = 0; j < 8; ++j) {
    int grow = rt * 16 + hv * 8 + j;       // global (s,b,t) row
    int s = grow / (B_ * T_);
    int rem = grow - s * (B_ * T_);
    int b = rem >> 9;
    int t = rem & (T_ - 1);
    float val = c[j] + bcol;
    size_t sbh = ((size_t)s * B_ + b) * NH_ + hh;
    if (sect == 0)      q[(sbh * T_ + t) * HD_ + dd] = (_Float16)val; // [sbh][t][d]
    else if (sect == 1) k[(sbh * HD_ + dd) * T_ + t] = (_Float16)val; // [sbh][d][t]
    else                v[(sbh * T_ + t) * HD_ + dd] = (_Float16)val; // [sbh][t][d]
  }
}

// ------------------------------- k_gru -------------------------------------
// One 128-thread workgroup per (seq,batch). h lives in LDS; each thread owns
// hidden unit e and gate rows {e, e+128, e+256} of w_hh (hot in L2/WGP$).
__global__ void __launch_bounds__(128)
k_gru(const float* __restrict__ a_in, const float* __restrict__ p_in,
      const float* __restrict__ n_in, const int* __restrict__ a_len,
      const int* __restrict__ p_len, const int* __restrict__ n_len,
      const float* __restrict__ w_ih, const float* __restrict__ w_hh,
      const float* __restrict__ b_ih, const float* __restrict__ b_hh,
      float* __restrict__ coor_out) {
  int sb = blockIdx.x;
  int s = sb >> 5, b = sb & 31;
  int e = threadIdx.x;
  const float* in   = (s == 0) ? a_in : (s == 1) ? p_in : n_in;
  const int*   lenp = (s == 0) ? a_len : (s == 1) ? p_len : n_len;
  int len = lenp[b];

  __shared__ float hs[E_];
  hs[e] = 0.f;

  float wr0 = w_ih[e * 2],         wr1 = w_ih[e * 2 + 1],         br = b_ih[e];
  float wz0 = w_ih[(e + 128) * 2], wz1 = w_ih[(e + 128) * 2 + 1], bz = b_ih[e + 128];
  float wn0 = w_ih[(e + 256) * 2], wn1 = w_ih[(e + 256) * 2 + 1], bn = b_ih[e + 256];
  float bhr = b_hh[e], bhz = b_hh[e + 128], bhn = b_hh[e + 256];
  const float* whr = w_hh + (size_t)e * E_;
  const float* whz = w_hh + (size_t)(e + 128) * E_;
  const float* whn = w_hh + (size_t)(e + 256) * E_;

  float hlast = 0.f;
  __syncthreads();
  for (int t = 0; t < T_; ++t) {
    float c0 = in[((size_t)b * T_ + t) * 4 + 0];
    float c1 = in[((size_t)b * T_ + t) * 4 + 1];
    float hr = bhr, hz = bhz, hn = bhn;
#pragma unroll 4
    for (int kk = 0; kk < E_; ++kk) {
      float hk = hs[kk];
      hr = fmaf(whr[kk], hk, hr);
      hz = fmaf(whz[kk], hk, hz);
      hn = fmaf(whn[kk], hk, hn);
    }
    float xr = fmaf(wr0, c0, fmaf(wr1, c1, br));
    float xz = fmaf(wz0, c0, fmaf(wz1, c1, bz));
    float xn = fmaf(wn0, c0, fmaf(wn1, c1, bn));
    float r  = 1.f / (1.f + __expf(-(xr + hr)));
    float z  = 1.f / (1.f + __expf(-(xz + hz)));
    float nn = tanhf(xn + r * hn);
    float h2 = (1.f - z) * nn + z * hs[e];
    __syncthreads();          // all reads of hs done
    hs[e] = h2;
    if (t == len - 1) hlast = h2;
    __syncthreads();          // new hs visible
  }
  coor_out[((size_t)s * B_ + b) * E_ + e] = hlast;
}

// ------------------------------- k_attn ------------------------------------
// One wave (=block) per (sbh, q-tile). f32 WMMA 16x16x4:
//   A-frag: lane L -> M=L&15, k pair = 2*(L>>4); C-frag: VGPR j -> row 8*hv+j,
//   col = L&15. Flash softmax with width-16 shuffles; P transposed via LDS.
__global__ void __launch_bounds__(32)
k_attn(const _Float16* __restrict__ q, const _Float16* __restrict__ k,
       const _Float16* __restrict__ v, float* __restrict__ part) {
  int blk  = blockIdx.x;
  int qt   = blk & 31;        // T/16 q-tiles
  int sbh  = blk >> 5;        // 0..1535
  int lane = threadIdx.x & 31;
  int M  = lane & 15;
  int hv = lane >> 4;

  const _Float16* qp = q + (size_t)sbh * T_ * HD_;
  const _Float16* kp = k + (size_t)sbh * HD_ * T_;
  const _Float16* vp = v + (size_t)sbh * T_ * HD_;

  // Q A-fragments for the two K=4 steps covering d=0..7.
  v2f qa0, qa1;
  {
    const _Float16* qrow = qp + (size_t)(qt * 16 + M) * HD_;
    int k0 = hv * 2;
    qa0[0] = (float)qrow[k0];     qa0[1] = (float)qrow[k0 + 1];
    qa1[0] = (float)qrow[4 + k0]; qa1[1] = (float)qrow[4 + k0 + 1];
  }

  float mrow[8], lrow[8];
  v8f o = {};
#pragma unroll
  for (int j = 0; j < 8; ++j) { mrow[j] = -1e30f; lrow[j] = 0.f; }

  __shared__ float pls[16][17];   // P transpose staging (pad vs bank conflicts)
  const float scale = 0.35355339059327373f;   // 1/sqrt(HD)

  for (int kt = 0; kt < 32; ++kt) {
    int kbase = kt * 16;
    if (kt + 1 < 32)  // gfx1250 global_prefetch_b8 on next K tile
      __builtin_prefetch(kp + (kbase + 16), 0, 3);

    // --- S = Q K^T : 2 chained K=4 f32 WMMAs (zero K padding, HD=8) ---
    v8f sfr = {};
    {
      int k0 = hv * 2;
      v2f kb0, kb1;   // B[k][n] = K_seq[t=kbase+n][d=k], k layout [d][t]
      kb0[0] = (float)kp[(size_t)(k0 + 0) * T_ + kbase + M];
      kb0[1] = (float)kp[(size_t)(k0 + 1) * T_ + kbase + M];
      kb1[0] = (float)kp[(size_t)(k0 + 4) * T_ + kbase + M];
      kb1[1] = (float)kp[(size_t)(k0 + 5) * T_ + kbase + M];
      sfr = __builtin_amdgcn_wmma_f32_16x16x4_f32(false, qa0, false, kb0,
                                                  (short)0, sfr, false, false);
      sfr = __builtin_amdgcn_wmma_f32_16x16x4_f32(false, qa1, false, kb1,
                                                  (short)0, sfr, false, false);
    }

    // --- streaming softmax update (per fragment row) ---
    float pj[8];
#pragma unroll
    for (int j = 0; j < 8; ++j) {
      float sv = sfr[j] * scale;
      float mx = sv;
      mx = fmaxf(mx, __shfl_xor(mx, 1, 16));
      mx = fmaxf(mx, __shfl_xor(mx, 2, 16));
      mx = fmaxf(mx, __shfl_xor(mx, 4, 16));
      mx = fmaxf(mx, __shfl_xor(mx, 8, 16));
      float mnew = fmaxf(mrow[j], mx);
      float corr = __expf(mrow[j] - mnew);
      float p    = __expf(sv - mnew);
      float rs = p;
      rs += __shfl_xor(rs, 1, 16);
      rs += __shfl_xor(rs, 2, 16);
      rs += __shfl_xor(rs, 4, 16);
      rs += __shfl_xor(rs, 8, 16);
      lrow[j] = lrow[j] * corr + rs;
      mrow[j] = mnew;
      o[j] *= corr;
      pj[j] = p;
    }

    // --- transpose P (C layout -> A layout) through LDS ---
    __syncthreads();
#pragma unroll
    for (int j = 0; j < 8; ++j) pls[hv * 8 + j][M] = pj[j];
    __syncthreads();

    // --- O += P V : 4 chained K=4 f32 WMMAs (K=16 exact) ---
#pragma unroll
    for (int ks = 0; ks < 4; ++ks) {
      int k0 = ks * 4 + hv * 2;
      v2f pa, vb;
      pa[0] = pls[M][k0];
      pa[1] = pls[M][k0 + 1];
      vb[0] = (M < HD_) ? (float)vp[(size_t)(kbase + k0) * HD_ + M] : 0.f;
      vb[1] = (M < HD_) ? (float)vp[(size_t)(kbase + k0 + 1) * HD_ + M] : 0.f;
      o = __builtin_amdgcn_wmma_f32_16x16x4_f32(false, pa, false, vb,
                                                (short)0, o, false, false);
    }
  }

  // Normalize rows by l, sum the 16 rows of this q-tile (mean-pool partial).
  float colsum = 0.f;
#pragma unroll
  for (int j = 0; j < 8; ++j) colsum += o[j] / lrow[j];
  colsum += __shfl_xor(colsum, 16, 32);   // combine row halves (same column)
  if (lane < HD_)                          // deterministic partials, no atomics
    part[((size_t)sbh * HD_ + lane) * 32 + qt] = colsum;
}

// ------------------------------ k_final ------------------------------------
// grid_emb = (mean_T o) @ out_w^T + out_b ; out = g*coor + (1-g)*grid_emb
__global__ void __launch_bounds__(128)
k_final(const float* __restrict__ part, const float* __restrict__ coor,
        const float* __restrict__ out_w, const float* __restrict__ out_b,
        const float* __restrict__ gamma_p, float* __restrict__ out) {
  int sb = blockIdx.x;            // 0..95  (s*B + b)
  int e  = threadIdx.x;           // 0..127
  __shared__ float arow[E_];
  {
    size_t base = (((size_t)sb * NH_ + (e >> 3)) * HD_ + (e & 7)) * 32;
    float ssum = 0.f;
#pragma unroll
    for (int qt = 0; qt < 32; ++qt) ssum += part[base + qt];
    arow[e] = ssum * (1.f / (float)T_);
  }
  __syncthreads();
  float dot = 0.f;
  const float* wr = out_w + (size_t)e * E_;
#pragma unroll 4
  for (int kk = 0; kk < E_; ++kk) dot = fmaf(wr[kk], arow[kk], dot);
  float g = gamma_p[0];
  out[(size_t)sb * E_ + e] =
      g * coor[(size_t)sb * E_ + e] + (1.f - g) * (dot + out_b[e]);
}

// ------------------------------- k_dist ------------------------------------
__global__ void __launch_bounds__(32) k_dist(float* __restrict__ out) {
  int b = threadIdx.x;
  if (b >= B_) return;
  float sp = 0.f, sn = 0.f;
  for (int e = 0; e < E_; ++e) {
    float a  = out[b * E_ + e];
    float pe = out[B_ * E_ + b * E_ + e];
    float ne = out[2 * B_ * E_ + b * E_ + e];
    float dp = a - pe + 1e-6f;
    float dn = a - ne + 1e-6f;
    sp = fmaf(dp, dp, sp);
    sn = fmaf(dn, dn, sn);
  }
  out[3 * B_ * E_ + b]      = __expf(-sqrtf(sp));
  out[3 * B_ * E_ + B_ + b] = __expf(-sqrtf(sn));
}

// ----------------------------- kernel_launch -------------------------------
extern "C" void kernel_launch(void* const* d_in, const int* in_sizes, int n_in,
                              void* d_out, int out_size, void* d_ws,
                              size_t ws_size, hipStream_t stream) {
  (void)in_sizes; (void)n_in; (void)out_size; (void)ws_size;
  const float* anc_in = (const float*)d_in[0];
  const float* pos_in = (const float*)d_in[1];
  const float* neg_in = (const float*)d_in[2];
  const int*   anc_len = (const int*)d_in[3];
  const int*   pos_len = (const int*)d_in[4];
  const int*   neg_len = (const int*)d_in[5];
  const float* gru_w_ih = (const float*)d_in[6];
  const float* gru_w_hh = (const float*)d_in[7];
  const float* gru_b_ih = (const float*)d_in[8];
  const float* gru_b_hh = (const float*)d_in[9];
  const float* lookup_w = (const float*)d_in[10];
  const float* lookup_b = (const float*)d_in[11];
  const float* posres_w = (const float*)d_in[12];
  const float* posres_b = (const float*)d_in[13];
  const float* attn_in_w = (const float*)d_in[14];
  const float* attn_in_b = (const float*)d_in[15];
  const float* attn_out_w = (const float*)d_in[16];
  const float* attn_out_b = (const float*)d_in[17];
  const float* gamma_p = (const float*)d_in[18];
  float* out = (float*)d_out;

  // Workspace carve-up (all offsets 256B aligned).
  const size_t NTOK = (size_t)NS_ * B_ * T_;        // 49152
  _Float16* m  = (_Float16*)d_ws;                   // NTOK*E halves
  _Float16* qb = m  + NTOK * E_;
  _Float16* kb = qb + NTOK * E_;
  _Float16* vb = kb + NTOK * E_;
  float* coor = (float*)(vb + NTOK * E_);           // NS*B*E
  float* part = coor + (size_t)NS_ * B_ * E_;       // NS*B*NH*HD*32

  {  // m featurizer
    int n = NS_ * B_ * T_ * E_;
    k_prep_m<<<(n + 255) / 256, 256, 0, stream>>>(
        anc_in, pos_in, neg_in, lookup_w, lookup_b, posres_w, posres_b, m);
  }
  {  // QKV projection: 3072 row-tiles x 24 col-tiles, 1 wave per tile
    int waves = ((NS_ * B_ * T_) / 16) * 24;
    k_qkv<<<waves / 8, 256, 0, stream>>>(m, attn_in_w, attn_in_b, qb, kb, vb);
  }
  k_gru<<<NS_ * B_, 128, 0, stream>>>(anc_in, pos_in, neg_in,
                                      anc_len, pos_len, neg_len,
                                      gru_w_ih, gru_w_hh, gru_b_ih, gru_b_hh,
                                      coor);
  k_attn<<<NS_ * B_ * NH_ * (T_ / 16), 32, 0, stream>>>(qb, kb, vb, part);
  k_final<<<NS_ * B_, 128, 0, stream>>>(part, coor, attn_out_w, attn_out_b,
                                        gamma_p, out);
  k_dist<<<1, 32, 0, stream>>>(out);
}
